// MultiHeadAttention_16930761081063
// MI455X (gfx1250) — matrix operand order repaired and checked by reading, not compiled, not run on hardware
//
#include <hip/hip_runtime.h>
#include <stdint.h>

#define HIDDEN 1024
#define HEADS  16
#define HEAD   64
#define T_SEQ  2048
#define BATCH  2

typedef __attribute__((ext_vector_type(16))) __bf16 v16bf;
typedef __attribute__((ext_vector_type(8)))  __bf16 v8bf;
typedef __attribute__((ext_vector_type(8)))  float  v8f;

union V16U { v16bf v; v8bf h[2]; };

__device__ __forceinline__ __bf16 f2bf(float f) {
  union { float f; uint32_t u; } x; x.f = f;
  uint32_t r = (x.u + 0x7FFFu + ((x.u >> 16) & 1u)) >> 16;  // RNE
  union { uint16_t s; __bf16 b; } o; o.s = (uint16_t)r;
  return o.b;
}

// gfx1250 async global->LDS copy (tracked by ASYNCcnt).
__device__ __forceinline__ void async_ld_b128(uint32_t ldsOff, const void* g) {
  asm volatile("global_load_async_to_lds_b128 %0, %1, off"
               :: "v"(ldsOff), "v"(g) : "memory");
}
#define WAIT_ASYNC(n) asm volatile("s_wait_asynccnt %0" :: "n"(n) : "memory")

__device__ __forceinline__ uint32_t lds_off(const void* p) {
  return (uint32_t)(uintptr_t)p;
}

// ---------------------------------------------------------------- convert ---
__global__ void k_f32_to_bf16(const float* __restrict__ src,
                              __bf16* __restrict__ dst, int n) {
  int i = blockIdx.x * blockDim.x + threadIdx.x;
  if (i < n) dst[i] = f2bf(src[i]);
}

// src [K,N] f32 row-major -> dst [N,K] bf16, 64x64 LDS-tiled (coalesced).
__global__ void __launch_bounds__(256)
k_transpose_bf16(const float* __restrict__ src, __bf16* __restrict__ dst,
                 int K, int N) {
  __shared__ __bf16 tile[64][66];
  const int tid = threadIdx.x;
  const int kt  = K >> 6;
  const int tk  = (blockIdx.x % kt) << 6;
  const int tn  = (blockIdx.x / kt) << 6;
  const int r   = tid >> 2;
  const int c4  = (tid & 3) << 4;
  const float* sp = src + (tk + r) * N + tn + c4;
#pragma unroll
  for (int i = 0; i < 16; ++i) tile[r][c4 + i] = f2bf(sp[i]);
  __syncthreads();
  const int nn = tid >> 2;
  const int kc = (tid & 3) << 4;
  __bf16* dp = dst + (tn + nn) * K + tk + kc;
  v8bf o0, o1;
#pragma unroll
  for (int i = 0; i < 8; ++i) {
    o0[i] = tile[kc + i][nn];
    o1[i] = tile[kc + 8 + i][nn];
  }
  *(v8bf*)dp = o0;
  *(v8bf*)(dp + 8) = o1;
}

// ---------------------------------------------------------- GEMM helpers ----
// Block = 8 waves, block tile 256x64, per-wave 32x64 (8 accumulators).
#define GEMM_STAGE(BUF, K0)                                                  \
  {                                                                          \
    const __bf16* ag_ = agp + (K0);                                          \
    async_ld_b128(aL[BUF],      ag_);                                        \
    async_ld_b128(aL[BUF] + 16, ag_ + 8);                                    \
    async_ld_b128(aL[BUF] + 32, ag_ + 16);                                   \
    async_ld_b128(aL[BUF] + 48, ag_ + 24);                                   \
    async_ld_b128(bL[BUF],      bgp + (K0));                                 \
  }

#define GEMM_COMPUTE(BUF)                                                    \
  {                                                                          \
    const __bf16* Ab_ = &Abuf[BUF][0];                                       \
    const __bf16* Bb_ = &Bbuf[BUF][0];                                       \
    V16U a0_, a1_;                                                           \
    const __bf16* ap_ = Ab_ + (wave * 32 + ll) * 32 + (lh << 3);             \
    a0_.h[0] = *(const v8bf*)ap_;                                            \
    a0_.h[1] = *(const v8bf*)(ap_ + 16);                                     \
    a1_.h[0] = *(const v8bf*)(ap_ + 512);                                    \
    a1_.h[1] = *(const v8bf*)(ap_ + 528);                                    \
    _Pragma("unroll")                                                        \
    for (int j = 0; j < 4; ++j) {                                            \
      V16U b_;                                                               \
      const __bf16* bp_ = Bb_ + (16 * j + ll) * 32 + (lh << 3);              \
      b_.h[0] = *(const v8bf*)bp_;                                           \
      b_.h[1] = *(const v8bf*)(bp_ + 16);                                    \
      acc0[j] = __builtin_amdgcn_wmma_f32_16x16x32_bf16(                     \
          false, a0_.v, false, b_.v, (short)0, acc0[j], false, false);       \
      acc1[j] = __builtin_amdgcn_wmma_f32_16x16x32_bf16(                     \
          false, a1_.v, false, b_.v, (short)0, acc1[j], false, false);       \
    }                                                                        \
  }

#define GEMM_MAIN_LOOP()                                                     \
  GEMM_STAGE(0, 0);                                                          \
  _Pragma("unroll 1")                                                        \
  for (int i = 0; i < NITER - 2; i += 2) {                                   \
    GEMM_STAGE(1, (i + 1) * 32); WAIT_ASYNC(5); __syncthreads();             \
    GEMM_COMPUTE(0); __syncthreads();                                        \
    GEMM_STAGE(0, (i + 2) * 32); WAIT_ASYNC(5); __syncthreads();             \
    GEMM_COMPUTE(1); __syncthreads();                                        \
  }                                                                          \
  GEMM_STAGE(1, (NITER - 1) * 32); WAIT_ASYNC(5); __syncthreads();           \
  GEMM_COMPUTE(0); __syncthreads();                                          \
  WAIT_ASYNC(0); __syncthreads();                                            \
  GEMM_COMPUTE(1);

// ------------------------------------------------------------- QKV GEMM -----
__global__ void __launch_bounds__(256)
k_qkv_gemm(const __bf16* __restrict__ xb, const __bf16* __restrict__ wt,
           const float* __restrict__ bias,
           __bf16* __restrict__ Qo, __bf16* __restrict__ Ko,
           __bf16* __restrict__ Vt) {
  __shared__ __bf16 Abuf[2][256 * 32];
  __shared__ __bf16 Bbuf[2][64 * 32];
  const int tid  = threadIdx.x;
  const int lane = tid & 31;
  const int wave = tid >> 5;
  const int lh = lane >> 4, ll = lane & 15;
  const int NT   = (3 * HIDDEN) / 64;
  const int mBlk = (blockIdx.x / NT) * 256;
  const int n0   = (blockIdx.x % NT) * 64;
  const int m0   = mBlk + wave * 32;

  const __bf16* agp = xb + (mBlk + tid) * HIDDEN;
  const int bcol = tid >> 2;
  const int bseg = (tid & 3) << 3;
  const __bf16* bgp = wt + (n0 + bcol) * HIDDEN + bseg;
  const uint32_t aL[2] = { lds_off(&Abuf[0][tid * 32]),
                           lds_off(&Abuf[1][tid * 32]) };
  const uint32_t bL[2] = { lds_off(&Bbuf[0][bcol * 32 + bseg]),
                           lds_off(&Bbuf[1][bcol * 32 + bseg]) };

  v8f acc0[4] = {}, acc1[4] = {};
  const int NITER = HIDDEN / 32;
  GEMM_MAIN_LOOP();

  const int rbase = lh << 3;
#pragma unroll
  for (int s = 0; s < 2; ++s) {
#pragma unroll
    for (int j = 0; j < 4; ++j) {
      const v8f* accp = s ? &acc1[j] : &acc0[j];
      int c = n0 + 16 * j + ll;
      float bv = bias[c];
      int cc = c & 1023;
      int h  = cc >> 6, d = cc & 63;
      int sel = c >> 10;                           // 0=Q 1=K 2=V
      int token0 = m0 + s * 16 + rbase;
      int bI = token0 >> 11;
      int t0 = token0 & 2047;
      int bh = bI * HEADS + h;
      if (sel == 2) {                              // V^T: contiguous in t
        v8bf pk;
#pragma unroll
        for (int v = 0; v < 8; ++v) pk[v] = f2bf((*accp)[v] + bv);
        *(v8bf*)(Vt + (bh * HEAD + d) * T_SEQ + t0) = pk;
      } else {
        __bf16* base = (sel == 0 ? Qo : Ko) + (bh * T_SEQ + t0) * HEAD + d;
#pragma unroll
        for (int v = 0; v < 8; ++v) base[v * HEAD] = f2bf((*accp)[v] + bv);
      }
    }
  }
}

// ----------------------------------------------------------- attention ------
// Block = 8 waves sharing one (b,h): wave w handles queries q0..q0+15.
// 32-key K/V tiles double-buffered in LDS via async copies; streaming
// (flash) softmax.  Output bf16 [B*T, HIDDEN].
__global__ void __launch_bounds__(256)
k_attention(const __bf16* __restrict__ Q, const __bf16* __restrict__ K,
            const __bf16* __restrict__ Vt, const float* __restrict__ mask,
            __bf16* __restrict__ out) {
  __shared__ __bf16 Kbuf[2][32 * 64];              // 4 KB per stage
  __shared__ __bf16 Vbuf[2][64 * 32];              // 4 KB per stage
  __shared__ __bf16 Pbuf[8][16 * 32];              // per-wave P staging
  const int tid  = threadIdx.x;
  const int lane = tid & 31;
  const int wave = tid >> 5;
  const int bh   = blockIdx.x >> 4;                // 32 bh
  const int q0   = ((blockIdx.x & 15) << 7) + (wave << 4);
  const int lh   = lane >> 4, ll = lane & 15;
  const int rbase = lh << 3;

  // async staging: K tile rows = keys (32x64), V^T tile rows = d (64x32)
  const int krow = tid >> 3, kseg = (tid & 7) << 3;
  const __bf16* Kg = K + (bh * T_SEQ + krow) * HEAD + kseg;
  const int vd = tid >> 2, vseg = (tid & 3) << 3;
  const __bf16* Vg = Vt + (bh * HEAD + vd) * T_SEQ + vseg;
  const uint32_t kL[2] = { lds_off(&Kbuf[0][krow * 64 + kseg]),
                           lds_off(&Kbuf[1][krow * 64 + kseg]) };
  const uint32_t vL[2] = { lds_off(&Vbuf[0][vd * 32 + vseg]),
                           lds_off(&Vbuf[1][vd * 32 + vseg]) };

  V16U aQ[2];
  const __bf16* qp = Q + (bh * T_SEQ + q0 + ll) * HEAD + (lh << 3);
  aQ[0].h[0] = *(const v8bf*)(qp);       aQ[0].h[1] = *(const v8bf*)(qp + 16);
  aQ[1].h[0] = *(const v8bf*)(qp + 32);  aQ[1].h[1] = *(const v8bf*)(qp + 48);

  v8f o[4] = {};
  float mrow[8], lrow[8];
#pragma unroll
  for (int v = 0; v < 8; ++v) { mrow[v] = -1e30f; lrow[v] = 0.f; }

  const float scale = 0.125f;                      // 1/sqrt(HEAD)
  __bf16* pbuf = Pbuf[wave];
  const float* maskRow = mask + (q0 + rbase) * T_SEQ + ll;

#define ATT_STAGE(BUF, S)                                                    \
  async_ld_b128(kL[BUF], Kg + (S) * 32 * HEAD);                              \
  async_ld_b128(vL[BUF], Vg + (S) * 32);

#define ATT_COMPUTE(BUF, KB)                                                 \
  {                                                                          \
    v8f s_[2] = {};                                                          \
    _Pragma("unroll")                                                        \
    for (int nt = 0; nt < 2; ++nt) {                                         \
      _Pragma("unroll")                                                      \
      for (int dc = 0; dc < 2; ++dc) {                                       \
        V16U b_;                                                             \
        const __bf16* kp_ =                                                  \
            &Kbuf[BUF][(16 * nt + ll) * 64 + dc * 32 + (lh << 3)];           \
        b_.h[0] = *(const v8bf*)kp_;                                         \
        b_.h[1] = *(const v8bf*)(kp_ + 16);                                  \
        s_[nt] = __builtin_amdgcn_wmma_f32_16x16x32_bf16(                    \
            false, aQ[dc].v, false, b_.v, (short)0, s_[nt], false, false);   \
      }                                                                      \
    }                                                                        \
    float mnew_[8];                                                          \
    _Pragma("unroll")                                                        \
    for (int v = 0; v < 8; ++v) {                                            \
      float mk0 = maskRow[v * T_SEQ + (KB)];                                 \
      float mk1 = maskRow[v * T_SEQ + (KB) + 16];                            \
      float s0 = (mk0 == 0.f) ? -1e30f : s_[0][v] * scale;                   \
      float s1 = (mk1 == 0.f) ? -1e30f : s_[1][v] * scale;                   \
      s_[0][v] = s0; s_[1][v] = s1;                                          \
      float mx = fmaxf(s0, s1);                                              \
      mx = fmaxf(mx, __shfl_xor(mx, 1, 32));                                 \
      mx = fmaxf(mx, __shfl_xor(mx, 2, 32));                                 \
      mx = fmaxf(mx, __shfl_xor(mx, 4, 32));                                 \
      mx = fmaxf(mx, __shfl_xor(mx, 8, 32));                                 \
      mnew_[v] = fmaxf(mrow[v], mx);                                         \
    }                                                                        \
    float alpha_[8];                                                         \
    _Pragma("unroll")                                                        \
    for (int v = 0; v < 8; ++v) {                                            \
      float p0 = __expf(s_[0][v] - mnew_[v]);                                \
      float p1 = __expf(s_[1][v] - mnew_[v]);                                \
      float ls = p0 + p1;                                                    \
      ls += __shfl_xor(ls, 1, 32);                                           \
      ls += __shfl_xor(ls, 2, 32);                                           \
      ls += __shfl_xor(ls, 4, 32);                                           \
      ls += __shfl_xor(ls, 8, 32);                                           \
      alpha_[v] = __expf(mrow[v] - mnew_[v]);                                \
      lrow[v]   = lrow[v] * alpha_[v] + ls;                                  \
      mrow[v]   = mnew_[v];                                                  \
      pbuf[(rbase + v) * 32 + ll]      = f2bf(p0);                           \
      pbuf[(rbase + v) * 32 + 16 + ll] = f2bf(p1);                           \
    }                                                                        \
    _Pragma("unroll")                                                        \
    for (int j = 0; j < 4; ++j)                                              \
      _Pragma("unroll")                                                      \
      for (int v = 0; v < 8; ++v) o[j][v] *= alpha_[v];                      \
    V16U aP_;                                                                \
    const __bf16* pp_ = pbuf + ll * 32 + (lh << 3);                          \
    aP_.h[0] = *(const v8bf*)(pp_);                                          \
    aP_.h[1] = *(const v8bf*)(pp_ + 16);                                     \
    _Pragma("unroll")                                                        \
    for (int j = 0; j < 4; ++j) {                                            \
      V16U b_;                                                               \
      const __bf16* vp_ = &Vbuf[BUF][(16 * j + ll) * 32 + (lh << 3)];        \
      b_.h[0] = *(const v8bf*)vp_;                                           \
      b_.h[1] = *(const v8bf*)(vp_ + 16);                                    \
      o[j] = __builtin_amdgcn_wmma_f32_16x16x32_bf16(                        \
          false, aP_.v, false, b_.v, (short)0, o[j], false, false);          \
    }                                                                        \
  }

  // 64 key-block stages, double-buffered, branch-free
  ATT_STAGE(0, 0);
#pragma unroll 1
  for (int i = 0; i < 62; i += 2) {
    ATT_STAGE(1, i + 1); WAIT_ASYNC(2); __syncthreads();
    ATT_COMPUTE(0, i * 32); __syncthreads();
    ATT_STAGE(0, i + 2); WAIT_ASYNC(2); __syncthreads();
    ATT_COMPUTE(1, (i + 1) * 32); __syncthreads();
  }
  ATT_STAGE(1, 63); WAIT_ASYNC(2); __syncthreads();
  ATT_COMPUTE(0, 62 * 32); __syncthreads();
  WAIT_ASYNC(0); __syncthreads();
  ATT_COMPUTE(1, 63 * 32);

  const int b = bh / HEADS, h = bh % HEADS;
#pragma unroll
  for (int j = 0; j < 4; ++j) {
    int d = 16 * j + ll;
#pragma unroll
    for (int v = 0; v < 8; ++v) {
      int t = q0 + rbase + v;
      float val = o[j][v] / lrow[v];
      out[(b * T_SEQ + t) * HIDDEN + h * HEAD + d] = f2bf(val);
    }
  }
#undef ATT_STAGE
#undef ATT_COMPUTE
}

// ------------------------------------------------------------ out proj ------
__global__ void __launch_bounds__(256)
k_proj_gemm(const __bf16* __restrict__ ab, const __bf16* __restrict__ wt,
            const float* __restrict__ bias, float* __restrict__ out) {
  __shared__ __bf16 Abuf[2][256 * 32];
  __shared__ __bf16 Bbuf[2][64 * 32];
  const int tid  = threadIdx.x;
  const int lane = tid & 31;
  const int wave = tid >> 5;
  const int lh = lane >> 4, ll = lane & 15;
  const int NT   = HIDDEN / 64;
  const int mBlk = (blockIdx.x / NT) * 256;
  const int n0   = (blockIdx.x % NT) * 64;
  const int m0   = mBlk + wave * 32;

  const __bf16* agp = ab + (mBlk + tid) * HIDDEN;
  const int bcol = tid >> 2;
  const int bseg = (tid & 3) << 3;
  const __bf16* bgp = wt + (n0 + bcol) * HIDDEN + bseg;
  const uint32_t aL[2] = { lds_off(&Abuf[0][tid * 32]),
                           lds_off(&Abuf[1][tid * 32]) };
  const uint32_t bL[2] = { lds_off(&Bbuf[0][bcol * 32 + bseg]),
                           lds_off(&Bbuf[1][bcol * 32 + bseg]) };

  v8f acc0[4] = {}, acc1[4] = {};
  const int NITER = HIDDEN / 32;
  GEMM_MAIN_LOOP();

  const int rbase = lh << 3;
#pragma unroll
  for (int s = 0; s < 2; ++s) {
#pragma unroll
    for (int j = 0; j < 4; ++j) {
      const v8f* accp = s ? &acc1[j] : &acc0[j];
      int c = n0 + 16 * j + ll;
      float bv = bias[c];
      int token0 = m0 + s * 16 + rbase;
#pragma unroll
      for (int v = 0; v < 8; ++v)
        out[(token0 + v) * HIDDEN + c] = (*accp)[v] + bv;
    }
  }
}

// ---------------------------------------------------------------- launch ----
extern "C" void kernel_launch(void* const* d_in, const int* in_sizes, int n_in,
                              void* d_out, int out_size, void* d_ws,
                              size_t ws_size, hipStream_t stream) {
  const float* x     = (const float*)d_in[0];
  const float* mask  = (const float*)d_in[1];
  const float* Wqkv  = (const float*)d_in[2];
  const float* bqkv  = (const float*)d_in[3];
  const float* Wproj = (const float*)d_in[4];
  const float* bproj = (const float*)d_in[5];
  float* out = (float*)d_out;

  char* ws = (char*)d_ws;
  __bf16* xb    = (__bf16*)(ws);                   //  8 MB  x as bf16
  __bf16* wqkvt = (__bf16*)(ws + 8388608);         //  6 MB  Wqkv^T bf16
  __bf16* wprjt = (__bf16*)(ws + 14680064);        //  2 MB  Wproj^T bf16
  __bf16* Qb    = (__bf16*)(ws + 16777216);        //  8 MB  Q [BH,T,D]
  __bf16* Kb    = (__bf16*)(ws + 25165824);        //  8 MB  K [BH,T,D]
  __bf16* Vtb   = (__bf16*)(ws + 33554432);        //  8 MB  V^T [BH,D,T]
  __bf16* attnb = (__bf16*)(ws + 41943040);        //  8 MB  attn out bf16

  k_f32_to_bf16<<<4194304 / 256, 256, 0, stream>>>(x, xb, 4194304);
  k_transpose_bf16<<<16 * 48, 256, 0, stream>>>(Wqkv, wqkvt, 1024, 3072);
  k_transpose_bf16<<<16 * 16, 256, 0, stream>>>(Wproj, wprjt, 1024, 1024);

  k_qkv_gemm<<<16 * 48, 256, 0, stream>>>(xb, wqkvt, bqkv, Qb, Kb, Vtb);
  k_attention<<<32 * 16, 256, 0, stream>>>(Qb, Kb, Vtb, mask, attnb);
  k_proj_gemm<<<16 * 16, 256, 0, stream>>>(attnb, wprjt, bproj, out);
}
